// VectorQuantizer_7155415515709
// MI455X (gfx1250) — matrix operand order repaired and checked by reading, not compiled, not run on hardware
//
#include <hip/hip_runtime.h>
#include <hip/hip_bf16.h>

typedef __attribute__((ext_vector_type(16))) _Float16 v16h;
typedef __attribute__((ext_vector_type(8)))  float    v8f;

#define KCODES 512
#define DCH    256
#define NROWS  65536          // 64*32*32 rows of z_flat
#define HWN    1024           // H*W
#define ZQ_ELEMS 16777216     // 64*256*1024

// workspace byte offsets (16B aligned)
#define WS_CB16   0           // 512*256*2   = 262144 B  (f16 codebook)
#define WS_EN     262144      // 512*4       =   2048 B  (||e||^2)
#define WS_IDX    264192      // 65536*4     = 262144 B  (int indices)
#define WS_LOSS   526336      // 4 B                     (sum of squared diffs)

#if defined(__has_builtin)
#if __has_builtin(__builtin_amdgcn_global_load_async_to_lds_b128)
#define VQ_ASYNC_LDS 1
#endif
#endif

// ---------------------------------------------------------------------------
// Kernel 0: codebook -> f16, per-code squared norms, zero loss accumulator
// ---------------------------------------------------------------------------
__global__ __launch_bounds__(256) void vq_prep(const float* __restrict__ cb,
                                               _Float16* __restrict__ cb16,
                                               float* __restrict__ en,
                                               float* __restrict__ loss) {
    __shared__ float s[256];
    int k = blockIdx.x;       // code id 0..511
    int d = threadIdx.x;      // channel 0..255
    float v = cb[k * DCH + d];
    cb16[k * DCH + d] = (_Float16)v;
    s[d] = v * v;
    __syncthreads();
    for (int off = 128; off > 0; off >>= 1) {
        if (d < off) s[d] += s[d + off];
        __syncthreads();
    }
    if (d == 0) en[k] = s[0];
    if (k == 0 && d == 0) *loss = 0.0f;
}

// ---------------------------------------------------------------------------
// Kernel 1: WMMA distance GEMM + argmin.
// 8 waves/block, 16 rows/wave -> 128 rows/block, 512 blocks.
// Full f16 codebook (256KB) + norms (2KB) staged in dynamic LDS.
// Inner loop: two independent accumulation chains (2 column tiles) sharing
// the same A fragments -> 16 WMMAs/iter with doubled ILP and halved LDS
// traffic per FLOP.
// ---------------------------------------------------------------------------
__global__ __launch_bounds__(256) void vq_argmin(const float* __restrict__ z,
                                                 const _Float16* __restrict__ cb16,
                                                 const float* __restrict__ en,
                                                 int* __restrict__ idx_out,
                                                 float* __restrict__ idx_f_out) {
    extern __shared__ __align__(32) char smem[];
    _Float16* cbS = (_Float16*)smem;               // 262144 B
    float*    enS = (float*)(smem + 262144);       //   2048 B

    // --- stage codebook into LDS ---
#ifdef VQ_ASYNC_LDS
    {
        // CDNA5 async copy: memory -> LDS without VGPR round-trip (ASYNCcnt).
        // Builtin signature (per hipcc diagnostic): arg0 = int4 addrspace(1)*
        // global source, arg1 = int4 addrspace(3)* LDS dest, imm offset, imm cpol.
        typedef __attribute__((ext_vector_type(4))) int i4_t;
        typedef __attribute__((address_space(1))) i4_t* gp_t;
        typedef __attribute__((address_space(3))) i4_t* lp_t;
        gp_t g = (gp_t)cb16;
        lp_t l = (lp_t)cbS;
        for (int t = threadIdx.x; t < (KCODES * DCH * 2) / 16; t += 256)
            __builtin_amdgcn_global_load_async_to_lds_b128(g + t, l + t, 0, 0);
        for (int t = threadIdx.x; t < KCODES; t += 256)
            enS[t] = en[t];
#if __has_builtin(__builtin_amdgcn_s_wait_asynccnt)
        __builtin_amdgcn_s_wait_asynccnt(0);
#else
        asm volatile("s_wait_asynccnt 0" ::: "memory");
#endif
    }
#else
    {
        const float4* src = (const float4*)cb16;
        float4*       dst = (float4*)cbS;
        for (int t = threadIdx.x; t < (KCODES * DCH * 2) / 16; t += 256)
            dst[t] = src[t];
        for (int t = threadIdx.x; t < KCODES; t += 256)
            enS[t] = en[t];
    }
#endif
    __syncthreads();

    const int lane     = threadIdx.x & 31;
    const int wave     = threadIdx.x >> 5;
    const int half_sel = lane >> 4;    // 0: lanes 0-15, 1: lanes 16-31
    const int m        = lane & 15;    // row within tile / column within tile

    const int row = blockIdx.x * 128 + wave * 16 + m;   // global z_flat row
    const int b   = row >> 10;
    const int hw  = row & (HWN - 1);
    const float* zrow = z + (size_t)b * (DCH * HWN) + hw;   // stride HWN over c
    __builtin_prefetch(zrow, 0, 0);

    // --- A fragments: 16 rows x 256 K, contiguous K order matching B:
    //     lanes 0-15 hold K=0..15, lanes 16-31 hold K=16..31
    //     (per 32-wide K-step) ---
    v16h afrag[8];
#pragma unroll
    for (int ks = 0; ks < 8; ++ks) {
#pragma unroll
        for (int h = 0; h < 16; ++h) {
            int K = h + half_sel * 16;
            afrag[ks][h] = (_Float16)zrow[(size_t)(ks * 32 + K) * HWN];
        }
    }

    float best[8];
    int   bidx[8];
#pragma unroll
    for (int r = 0; r < 8; ++r) { best[r] = 3.4e38f; bidx[r] = 0; }

    // --- sweep 32 column tiles of 16 codes, two tiles per iteration ---
    for (int ct = 0; ct < 32; ct += 2) {
        const int code0 = ct * 16 + m;                 // this lane's columns
        const int code1 = code0 + 16;
        const _Float16* brow0 = cbS + code0 * DCH;
        const _Float16* brow1 = cbS + code1 * DCH;
        v8f acc0 = {};
        v8f acc1 = {};
#pragma unroll
        for (int ks = 0; ks < 8; ++ks) {
            // B 32x16 f16 layout: lanes 0-15 K=0..15, lanes 16-31 K=16..31,
            // contiguous 32B per lane -> two ds_load_b128 each
            v16h b0 = *(const v16h*)(brow0 + ks * 32 + half_sel * 16);
            v16h b1 = *(const v16h*)(brow1 + ks * 32 + half_sel * 16);
            acc0 = __builtin_amdgcn_wmma_f32_16x16x32_f16(
                false, afrag[ks], false, b0, (short)0, acc0, false, false);
            acc1 = __builtin_amdgcn_wmma_f32_16x16x32_f16(
                false, afrag[ks], false, b1, (short)0, acc1, false, false);
        }
        float en0 = enS[code0];                        // ||e||^2 per column
        float en1 = enS[code1];
#pragma unroll
        for (int r = 0; r < 8; ++r) {
            float d0 = en0 - 2.0f * acc0[r];           // ||z||^2 is row-constant
            if (d0 < best[r]) { best[r] = d0; bidx[r] = code0; }
            float d1 = en1 - 2.0f * acc1[r];
            if (d1 < best[r]) { best[r] = d1; bidx[r] = code1; }
        }
    }

    // --- per-row argmin: reduce across the 16 lanes of each half-wave ---
#pragma unroll
    for (int r = 0; r < 8; ++r) {
        float bv = best[r];
        int   bi = bidx[r];
        for (int msk = 1; msk < 16; msk <<= 1) {
            float ov = __shfl_xor(bv, msk, 32);
            int   oi = __shfl_xor(bi, msk, 32);
            if (ov < bv || (ov == bv && oi < bi)) { bv = ov; bi = oi; }
        }
        if (m == 0) {   // lanes 0 (rows 0..7) and 16 (rows 8..15)
            int orow = blockIdx.x * 128 + wave * 16 + half_sel * 8 + r;
            idx_out[orow]   = bi;
            idx_f_out[orow] = (float)bi;
        }
    }
}

// ---------------------------------------------------------------------------
// Kernel 2: gather + straight-through output + loss partials
// ---------------------------------------------------------------------------
__global__ __launch_bounds__(256) void vq_gather(const float* __restrict__ z,
                                                 const float* __restrict__ cb,
                                                 const int* __restrict__ idx,
                                                 float* __restrict__ out,
                                                 float* __restrict__ loss) {
    __shared__ float s[256];
    const int blk = blockIdx.x;
    const int hw  = (blk & 3) * 256 + threadIdx.x;
    const int c   = (blk >> 2) & 255;
    const int b   = blk >> 10;
    const int i   = (b << 10) | hw;                    // z_flat row
    const size_t e = (size_t)b * (DCH * HWN) + (size_t)c * HWN + hw;

    const int k   = idx[i];
    const float ze = z[e];
    const float zq = cb[k * DCH + c];
    out[e] = ze + (zq - ze);                            // straight-through value
    const float diff = ze - zq;
    s[threadIdx.x] = diff * diff;
    __syncthreads();
    for (int off = 128; off > 0; off >>= 1) {
        if ((int)threadIdx.x < off) s[threadIdx.x] += s[threadIdx.x + off];
        __syncthreads();
    }
    if (threadIdx.x == 0) atomicAdd(loss, s[0]);
}

// ---------------------------------------------------------------------------
// Kernel 3: finalize both loss scalars (identical means)
// ---------------------------------------------------------------------------
__global__ void vq_finalize(const float* __restrict__ loss,
                            float* __restrict__ out_losses) {
    float mean = *loss / (float)ZQ_ELEMS;
    out_losses[0] = mean;   // commitment_loss
    out_losses[1] = mean;   // codebook_loss
}

// ---------------------------------------------------------------------------
extern "C" void kernel_launch(void* const* d_in, const int* in_sizes, int n_in,
                              void* d_out, int out_size, void* d_ws, size_t ws_size,
                              hipStream_t stream) {
    const float* z_e = (const float*)d_in[0];   // (64,256,32,32) f32
    const float* cb  = (const float*)d_in[1];   // (512,256) f32
    float* out = (float*)d_out;
    char*  ws  = (char*)d_ws;

    _Float16* cb16 = (_Float16*)(ws + WS_CB16);
    float*    en   = (float*)(ws + WS_EN);
    int*      idxb = (int*)(ws + WS_IDX);
    float*    loss = (float*)(ws + WS_LOSS);

    vq_prep<<<KCODES, 256, 0, stream>>>(cb, cb16, en, loss);
    vq_argmin<<<NROWS / 128, 256, 264192, stream>>>(z_e, cb16, en, idxb,
                                                    out + ZQ_ELEMS + 2);
    vq_gather<<<ZQ_ELEMS / 256, 256, 0, stream>>>(z_e, cb, idxb, out, loss);
    vq_finalize<<<1, 1, 0, stream>>>(loss, out + ZQ_ELEMS);
}
